// Cross_Frame_Att_17755394801887
// MI455X (gfx1250) — matrix operand order repaired and checked by reading, not compiled, not run on hardware
//
#include <hip/hip_runtime.h>
#include <math.h>

// ---------------- types ----------------
typedef __bf16 bf16;
typedef bf16  v16bf __attribute__((ext_vector_type(16)));
typedef bf16  v8bf  __attribute__((ext_vector_type(8)));
typedef float v8f   __attribute__((ext_vector_type(8)));

#define DEVFN __device__ __forceinline__

// ---------------- problem constants ----------------
constexpr int B = 8, F = 4, C = 128, N = 1024, H = 4;
constexpr int DQ  = 4 * C;   // 512
constexpr int DKV = 8 * C;   // 1024
constexpr float EPS   = 1e-5f;
constexpr float SCALE = 0.08838834764831845f;  // 1/sqrt(128)

// ---------------- workspace layout (bytes) ----------------
constexpr size_t WS_SS   = 0;                                        // scale/shift f32 [B*C*2]
constexpr size_t WS_XNB  = WS_SS   + (size_t)2 * B * C * 4;          // xn bf16 (b,f,n,c)
constexpr size_t WS_WQ   = WS_XNB  + (size_t)B * F * N * C * 2;      // Wq bf16
constexpr size_t WS_WKV  = WS_WQ   + (size_t)DQ * C * 2;             // Wkv bf16
constexpr size_t WS_WP   = WS_WKV  + (size_t)DKV * C * 2;            // Wp bf16
constexpr size_t WS_Q    = WS_WP   + (size_t)C * C * 2;              // q bf16 (bf,n,512)
constexpr size_t WS_KV   = WS_Q    + (size_t)B * 2 * N * DQ * 2;     // kv bf16 (bf,n,1024)
constexpr size_t WS_VT   = WS_KV   + (size_t)B * 2 * N * DKV * 2;    // v^T bf16 (bfh,c,n)
constexpr size_t WS_OV   = WS_VT   + (size_t)B * 2 * H * C * N * 2;  // ov f32 (bf,h,n,c)
constexpr size_t WS_XA   = WS_OV   + (size_t)B * 2 * H * N * C * 4;  // xa f32 (b,h,n,c)
constexpr size_t WS_PRED = WS_XA   + (size_t)B * H * N * C * 4;      // pred f32 (b,n,3)
constexpr size_t WS_GT   = WS_PRED + (size_t)B * N * 3 * 4;          // gt f32 (b,n,3)
constexpr size_t WS_PART = WS_GT   + (size_t)B * N * 3 * 4;          // 64 partials

// ---------------- WMMA helpers (CDNA5 bf16 16x16x32, f32 acc) ----------------
DEVFN v8f wmma_bf16(v16bf a, v16bf b, v8f c) {
  return __builtin_amdgcn_wmma_f32_16x16x32_bf16(
      /*neg_a=*/false, a, /*neg_b=*/false, b,
      /*c_mod=*/(short)0, c, /*reuse_a=*/false, /*reuse_b=*/false);
}

// two 8-element bf16 runs -> one 16-element fragment
DEVFN v16bf frag2x8(const bf16* p0, const bf16* p1) {
  v8bf lo = *(const v8bf*)p0;
  v8bf hi = *(const v8bf*)p1;
  v16bf r;
#pragma unroll
  for (int i = 0; i < 8; ++i) { r[i] = lo[i]; r[i + 8] = hi[i]; }
  return r;
}
// A-frag: row-major A[m][k], lane row m, k = k0 + 8*half + {0..7} and +16
DEVFN v16bf load_a_bf16(const bf16* A, int lda, int m, int k0, int half) {
  const bf16* row = A + (size_t)m * lda + k0 + half * 8;
  return frag2x8(row, row + 16);
}
// B-frag from row-major W[nd][k] (B[k][nd] = W[nd][k]): lane row nd, k = k0 + 8*half + {0..7} and +16
DEVFN v16bf load_b_bf16(const bf16* Wrow, int k0, int half) {
  const bf16* p = Wrow + k0 + half * 8;
  return frag2x8(p, p + 16);
}

// ================= kernel 1: BN stats -> fused scale/shift =================
__global__ void k_bn_stats(const float* __restrict__ xs, const float* __restrict__ gamma,
                           const float* __restrict__ beta, float* __restrict__ ss) {
  int bc = blockIdx.x;           // b*C + c
  int b = bc >> 7, c = bc & 127;
  const float* base = xs + ((size_t)(b * F) * C + c) * N;
  float s = 0.f, sq = 0.f;
  for (int f = 0; f < F; ++f) {
    const float* p = base + (size_t)f * C * N;
    for (int n = threadIdx.x; n < N; n += 128) { float v = p[n]; s += v; sq += v * v; }
  }
  __shared__ float rs[128], rq[128];
  rs[threadIdx.x] = s; rq[threadIdx.x] = sq;
  __syncthreads();
  for (int st = 64; st > 0; st >>= 1) {
    if ((int)threadIdx.x < st) { rs[threadIdx.x] += rs[threadIdx.x + st]; rq[threadIdx.x] += rq[threadIdx.x + st]; }
    __syncthreads();
  }
  if (threadIdx.x == 0) {
    float inv_n = 1.f / (float)(F * N);
    float mean = rs[0] * inv_n;
    float var  = rq[0] * inv_n - mean * mean;
    float istd = rsqrtf(var + EPS);
    float scl  = istd * gamma[c];
    ss[bc * 2 + 0] = scl;
    ss[bc * 2 + 1] = beta[c] - mean * scl;
  }
}

// ============ kernel 2: normalize + (C,N)->(N,C) transpose -> bf16 ============
__global__ void k_norm_t(const float* __restrict__ xs, const float* __restrict__ ss,
                         bf16* __restrict__ xnb) {
  int bf = blockIdx.x;                    // b*F + f
  int b  = bf / F;
  int n0 = blockIdx.y * 64;
  __shared__ float t[128 * 65];
  const float* src = xs + (size_t)bf * C * N;
#pragma unroll
  for (int i = 0; i < 32; ++i) {
    int idx = i * 256 + threadIdx.x;      // 8192 = 128c x 64n
    int c = idx >> 6, nl = idx & 63;
    float scl = ss[(b * C + c) * 2], sh = ss[(b * C + c) * 2 + 1];
    t[c * 65 + nl] = src[(size_t)c * N + n0 + nl] * scl + sh;
  }
  __syncthreads();
  bf16* dst = xnb + ((size_t)bf * N + n0) * C;
#pragma unroll
  for (int i = 0; i < 32; ++i) {
    int idx = i * 256 + threadIdx.x;
    int nl = idx >> 7, c = idx & 127;
    dst[(size_t)nl * C + c] = (bf16)t[c * 65 + nl];
  }
}

// ================= kernel 3: f32 -> bf16 weight convert =================
__global__ void k_cvt_bf16(const float* __restrict__ in, bf16* __restrict__ out, int n) {
  int i = blockIdx.x * 256 + threadIdx.x;
  if (i < n) out[i] = (bf16)in[i];
}

// ========== kernel 4: q/kv projections (WMMA), f in {0,1}; kv from x[3-f] ==========
__global__ void k_qkv(const bf16* __restrict__ xnb, const bf16* __restrict__ Wq,
                      const bf16* __restrict__ Wkv, bf16* __restrict__ qo,
                      bf16* __restrict__ kvo) {
  int bf = blockIdx.x;                    // b*2 + f
  int b = bf >> 1, f = bf & 1;
  int m0 = blockIdx.y * 16;
  int lane = threadIdx.x & 31, wave = threadIdx.x >> 5, half = lane >> 4;
  int mrow = m0 + (lane & 15);

  const bf16* Aq  = xnb + ((size_t)(b * F + f)       * N) * C;
  const bf16* Akv = xnb + ((size_t)(b * F + (3 - f)) * N) * C;

  v16bf aq[4], akv[4];
#pragma unroll
  for (int ks = 0; ks < 4; ++ks) {
    aq[ks]  = load_a_bf16(Aq,  C, mrow, ks * 32, half);
    akv[ks] = load_a_bf16(Akv, C, mrow, ks * 32, half);
  }
  bf16* qrow  = qo  + ((size_t)bf * N + m0) * DQ;
  bf16* kvrow = kvo + ((size_t)bf * N + m0) * DKV;

  for (int nd = wave; nd < DQ / 16; nd += 8) {
    v8f acc = {};
    const bf16* w = Wq + (size_t)(nd * 16 + (lane & 15)) * C;
#pragma unroll
    for (int ks = 0; ks < 4; ++ks) acc = wmma_bf16(aq[ks], load_b_bf16(w, ks * 32, half), acc);
#pragma unroll
    for (int r = 0; r < 8; ++r)
      qrow[(size_t)(r + 8 * half) * DQ + nd * 16 + (lane & 15)] = (bf16)acc[r];
  }
  for (int nd = wave; nd < DKV / 16; nd += 8) {
    v8f acc = {};
    const bf16* w = Wkv + (size_t)(nd * 16 + (lane & 15)) * C;
#pragma unroll
    for (int ks = 0; ks < 4; ++ks) acc = wmma_bf16(akv[ks], load_b_bf16(w, ks * 32, half), acc);
#pragma unroll
    for (int r = 0; r < 8; ++r)
      kvrow[(size_t)(r + 8 * half) * DKV + nd * 16 + (lane & 15)] = (bf16)acc[r];
  }
}

// ================= kernel 5: V -> V^T (c,n) per (bf,h) =================
__global__ void k_vt(const bf16* __restrict__ kvo, bf16* __restrict__ vt) {
  int bfh = blockIdx.x;                   // (b*2+f)*H + h
  int n0  = blockIdx.y * 64;
  int h = bfh & 3, bf = bfh >> 2;
  __shared__ bf16 t[128 * 65];
  const bf16* src = kvo + ((size_t)bf * N + n0) * DKV + 512 + h * 128;
#pragma unroll
  for (int i = 0; i < 32; ++i) {
    int idx = i * 256 + threadIdx.x;
    int nl = idx >> 7, c = idx & 127;
    t[c * 65 + nl] = src[(size_t)nl * DKV + c];
  }
  __syncthreads();
  bf16* dst = vt + ((size_t)bfh * C) * N + n0;
#pragma unroll
  for (int i = 0; i < 32; ++i) {
    int idx = i * 256 + threadIdx.x;
    int c = idx >> 6, nl = idx & 63;
    dst[(size_t)c * N + nl] = t[c * 65 + nl];
  }
}

// ===== kernel 6: attention per (bf,h): 64-row q block =====
// Score panel lives in LDS as bf16 (128KB). Softmax stores UNNORMALIZED exp in
// place (same element size -> no in-place hazard); 1/rowsum is applied to the
// f32 O-accumulator at store time. Phase-3 A-frags are direct b128 LDS loads.
__global__ void __launch_bounds__(128)
k_attn(const bf16* __restrict__ qo, const bf16* __restrict__ kvo,
       const bf16* __restrict__ vt, float* __restrict__ ov) {
  extern __shared__ bf16 Sb[];            // 64 rows x 1024 cols bf16 = 128KB
  __shared__ float red[128];
  __shared__ float inv_s[64];

  int bfh = blockIdx.x;
  int bf = bfh >> 2, h = bfh & 3;
  int m0 = blockIdx.y * 64;
  int lane = threadIdx.x & 31, wave = threadIdx.x >> 5, half = lane >> 4;

  const bf16* qp = qo  + ((size_t)bf * N) * DQ  + h * C;   // + n*DQ
  const bf16* kp = kvo + ((size_t)bf * N) * DKV + h * C;   // + n'*DKV  (k part)
  int mrow = m0 + wave * 16 + (lane & 15);

  v16bf aq[4];
#pragma unroll
  for (int ks = 0; ks < 4; ++ks) aq[ks] = load_a_bf16(qp, DQ, mrow, ks * 32, half);

  // phase 1: S = q k^T  (per wave: 16 rows x all 1024 keys), store bf16
  for (int kt = 0; kt < 64; ++kt) {
    v8f acc = {};
    const bf16* krow = kp + (size_t)(kt * 16 + (lane & 15)) * DKV;
#pragma unroll
    for (int ks = 0; ks < 4; ++ks) acc = wmma_bf16(aq[ks], load_b_bf16(krow, ks * 32, half), acc);
#pragma unroll
    for (int r = 0; r < 8; ++r)
      Sb[(size_t)(wave * 16 + r + 8 * half) * 1024 + kt * 16 + (lane & 15)] = (bf16)acc[r];
  }
  __syncthreads();

  // phase 2: row max + unnormalized exp (2 threads per row, b128 chunks)
  {
    int row = threadIdx.x >> 1, part = threadIdx.x & 1;
    bf16* rp = Sb + (size_t)row * 1024 + part * 512;
    float mx = -3.4e38f;
    for (int jb = 0; jb < 64; ++jb) {
      v8bf v = *(const v8bf*)(rp + jb * 8);
#pragma unroll
      for (int i = 0; i < 8; ++i) mx = fmaxf(mx, (float)v[i]);
    }
    red[threadIdx.x] = mx;
    __syncthreads();
    mx = fmaxf(red[row * 2], red[row * 2 + 1]);
    float sum = 0.f;
    for (int jb = 0; jb < 64; ++jb) {
      v8bf v = *(const v8bf*)(rp + jb * 8);
      v8bf e;
#pragma unroll
      for (int i = 0; i < 8; ++i) {
        float x = __expf(((float)v[i] - mx) * SCALE);
        sum += x; e[i] = (bf16)x;
      }
      *(v8bf*)(rp + jb * 8) = e;
    }
    __syncthreads();                      // red[] reads (mx) done before reuse
    red[threadIdx.x] = sum;
    __syncthreads();
    if (part == 0) inv_s[row] = 1.f / (red[row * 2] + red[row * 2 + 1]);
  }
  __syncthreads();

  // phase 3: O = exp(S) @ V, scaled by 1/rowsum at store
  float myinv[8];
#pragma unroll
  for (int r = 0; r < 8; ++r) myinv[r] = inv_s[wave * 16 + r + 8 * half];
  const bf16* vtp = vt + ((size_t)bfh * C) * N;
  float* orow = ov + ((size_t)bfh * N + m0 + wave * 16) * C;
  const bf16* Pb = Sb + (size_t)(wave * 16 + (lane & 15)) * 1024;
  for (int ct = 0; ct < 8; ++ct) {
    v8f acc = {};
    const bf16* vrow = vtp + (size_t)(ct * 16 + (lane & 15)) * N;
    for (int ks2 = 0; ks2 < 32; ++ks2) {
      int kk0 = ks2 * 32;
      const bf16* pa = Pb + kk0 + half * 8;
      acc = wmma_bf16(frag2x8(pa, pa + 16), load_b_bf16(vrow, kk0, half), acc);
    }
#pragma unroll
    for (int r = 0; r < 8; ++r)
      orow[(size_t)(r + 8 * half) * C + ct * 16 + (lane & 15)] = acc[r] * myinv[r];
  }
}

// ===== kernel 7: xa = (ov_f0 + ov_f1) @ Wp^T + bp  (WMMA) =====
__global__ void k_proj(const float* __restrict__ ov, const bf16* __restrict__ Wp,
                       const float* __restrict__ bp, float* __restrict__ xa) {
  int bh = blockIdx.x;                    // b*H + h
  int b = bh >> 2, h = bh & 3;
  int m0 = blockIdx.y * 16;
  int lane = threadIdx.x & 31, wave = threadIdx.x >> 5, half = lane >> 4;
  const float* o0 = ov + (((size_t)(b * 2 + 0) * H + h) * N + m0 + (lane & 15)) * C;
  const float* o1 = ov + (((size_t)(b * 2 + 1) * H + h) * N + m0 + (lane & 15)) * C;

  v16bf a[4];
#pragma unroll
  for (int ks = 0; ks < 4; ++ks) {
    int k0 = ks * 32 + half * 8;
#pragma unroll
    for (int i = 0; i < 8; ++i) {
      a[ks][i]     = (bf16)(o0[k0 + i]      + o1[k0 + i]);
      a[ks][i + 8] = (bf16)(o0[k0 + 16 + i] + o1[k0 + 16 + i]);
    }
  }
  int nd = wave;                          // 8 waves cover C/16 = 8 tiles
  v8f acc = {};
  const bf16* w = Wp + (size_t)(nd * 16 + (lane & 15)) * C;
#pragma unroll
  for (int ks = 0; ks < 4; ++ks) acc = wmma_bf16(a[ks], load_b_bf16(w, ks * 32, half), acc);
  float bias = bp[nd * 16 + (lane & 15)];
  float* xrow = xa + ((size_t)bh * N + m0) * C;
#pragma unroll
  for (int r = 0; r < 8; ++r)
    xrow[(size_t)(r + 8 * half) * C + nd * 16 + (lane & 15)] = acc[r] + bias;
}

// ===== kernel 8: frames head; h==0 -> pred/gt, h>=1 -> output frames =====
__global__ void k_frames(const float* __restrict__ xa, const float* __restrict__ Wm,
                         const float* __restrict__ bm, const float* __restrict__ pc1,
                         float* __restrict__ out_frames, float* __restrict__ pred,
                         float* __restrict__ gt) {
  int idx = blockIdx.x * 256 + threadIdx.x;   // over B*H*N
  if (idx >= B * H * N) return;
  int n = idx & (N - 1);
  int bh = idx >> 10;
  int h = bh & 3, b = bh >> 2;
  const float* row = xa + (size_t)idx * C;
  float f0 = bm[0], f1 = bm[1], f2 = bm[2];
  for (int c = 0; c < C; ++c) {
    float v = row[c];
    f0 += v * Wm[0 * C + c]; f1 += v * Wm[1 * C + c]; f2 += v * Wm[2 * C + c];
  }
  if (h == 0) {
    float g0 = pc1[((size_t)b * 3 + 0) * N + n];
    float g1 = pc1[((size_t)b * 3 + 1) * N + n];
    float g2 = pc1[((size_t)b * 3 + 2) * N + n];
    size_t o = ((size_t)b * N + n) * 3;
    pred[o] = g0 + f0; pred[o + 1] = g1 + f1; pred[o + 2] = g2 + f2;
    gt[o] = g0; gt[o + 1] = g1; gt[o + 2] = g2;
  } else {
    size_t o = (((size_t)b * 3 + (h - 1)) * N + n) * 3;
    out_frames[o] = f0; out_frames[o + 1] = f1; out_frames[o + 2] = f2;
  }
}

// ===== kernel 9: copy xa heads 1..3 into d_out =====
__global__ void k_copy_xa(const float* __restrict__ xa, float* __restrict__ out) {
  size_t idx = (size_t)blockIdx.x * 256 + threadIdx.x;
  if (idx >= (size_t)B * 3 * N * C) return;
  int c = (int)(idx & (C - 1));
  size_t r = idx >> 7;
  int n = (int)(r & (N - 1));
  size_t r2 = r >> 10;
  int hm = (int)(r2 % 3), b = (int)(r2 / 3);
  out[idx] = xa[(((size_t)b * H + (hm + 1)) * N + n) * C + c];
}

// ===== kernel 10: chamfer one direction, deterministic block partials =====
__global__ void k_chamfer(const float* __restrict__ P, const float* __restrict__ Q,
                          float* __restrict__ partial, int poff) {
  int b = blockIdx.x, chunk = blockIdx.y;
  __shared__ float q[N * 3];
  const float* Qb = Q + (size_t)b * N * 3;
  for (int i = threadIdx.x; i < N * 3; i += 256) q[i] = Qb[i];
  __syncthreads();
  int n = chunk * 256 + threadIdx.x;
  const float* p = P + ((size_t)b * N + n) * 3;
  float px = p[0], py = p[1], pz = p[2];
  float mn = 3.4e38f;
  for (int j = 0; j < N; ++j) {
    float dx = px - q[j * 3], dy = py - q[j * 3 + 1], dz = pz - q[j * 3 + 2];
    mn = fminf(mn, dx * dx + dy * dy + dz * dz);
  }
  __shared__ float red[256];
  red[threadIdx.x] = mn;
  __syncthreads();
  for (int st = 128; st > 0; st >>= 1) {
    if ((int)threadIdx.x < st) red[threadIdx.x] += red[threadIdx.x + st];
    __syncthreads();
  }
  if (threadIdx.x == 0) partial[poff + b * 4 + chunk] = red[0];
}

// ===== kernel 11: final loss reduction (single thread, deterministic) =====
__global__ void k_loss(const float* __restrict__ partial, float* __restrict__ out) {
  float s = 0.f;
  for (int i = 0; i < 64; ++i) s += partial[i];
  out[0] = s / (float)(B * N);
}

// ================================ launch ================================
extern "C" void kernel_launch(void* const* d_in, const int* in_sizes, int n_in,
                              void* d_out, int out_size, void* d_ws, size_t ws_size,
                              hipStream_t stream) {
  const float* xs    = (const float*)d_in[0];
  const float* pc1   = (const float*)d_in[1];
  // pc2 = d_in[2] unused (also unused in reference)
  const float* gamma = (const float*)d_in[3];
  const float* beta  = (const float*)d_in[4];
  const float* Wq    = (const float*)d_in[5];
  const float* Wkv   = (const float*)d_in[6];
  const float* Wp    = (const float*)d_in[7];
  const float* bp    = (const float*)d_in[8];
  const float* Wm    = (const float*)d_in[9];
  const float* bm    = (const float*)d_in[10];

  char* ws = (char*)d_ws;
  float* ss    = (float*)(ws + WS_SS);
  bf16*  xnb   = (bf16*)(ws + WS_XNB);
  bf16*  wq_b  = (bf16*)(ws + WS_WQ);
  bf16*  wkv_b = (bf16*)(ws + WS_WKV);
  bf16*  wp_b  = (bf16*)(ws + WS_WP);
  bf16*  qbf   = (bf16*)(ws + WS_Q);
  bf16*  kvbf  = (bf16*)(ws + WS_KV);
  bf16*  vtb   = (bf16*)(ws + WS_VT);
  float* ovf   = (float*)(ws + WS_OV);
  float* xaf   = (float*)(ws + WS_XA);
  float* pred  = (float*)(ws + WS_PRED);
  float* gt    = (float*)(ws + WS_GT);
  float* part  = (float*)(ws + WS_PART);

  float* out      = (float*)d_out;
  float* out_frm  = out + (size_t)B * 3 * N * C;
  float* out_loss = out + out_size - 1;

  const int attn_lds = 64 * 1024 * 2;  // 128KB bf16 score panel (<=320KB/WG on CDNA5)
  (void)hipFuncSetAttribute((const void*)k_attn, hipFuncAttributeMaxDynamicSharedMemorySize, attn_lds);

  k_bn_stats<<<dim3(B * C), dim3(128), 0, stream>>>(xs, gamma, beta, ss);
  k_norm_t  <<<dim3(B * F, N / 64), dim3(256), 0, stream>>>(xs, ss, xnb);
  k_cvt_bf16<<<dim3((DQ * C + 255) / 256),  dim3(256), 0, stream>>>(Wq,  wq_b,  DQ * C);
  k_cvt_bf16<<<dim3((DKV * C + 255) / 256), dim3(256), 0, stream>>>(Wkv, wkv_b, DKV * C);
  k_cvt_bf16<<<dim3((C * C + 255) / 256),   dim3(256), 0, stream>>>(Wp,  wp_b,  C * C);
  k_qkv <<<dim3(B * 2, N / 16), dim3(256), 0, stream>>>(xnb, wq_b, wkv_b, qbf, kvbf);
  k_vt  <<<dim3(B * 2 * H, N / 64), dim3(256), 0, stream>>>(kvbf, vtb);
  k_attn<<<dim3(B * 2 * H, N / 64), dim3(128), attn_lds, stream>>>(qbf, kvbf, vtb, ovf);
  k_proj<<<dim3(B * H, N / 16), dim3(256), 0, stream>>>(ovf, wp_b, bp, xaf);
  k_frames <<<dim3((B * H * N + 255) / 256), dim3(256), 0, stream>>>(xaf, Wm, bm, pc1, out_frm, pred, gt);
  k_copy_xa<<<dim3((B * 3 * N * C + 255) / 256), dim3(256), 0, stream>>>(xaf, out);
  k_chamfer<<<dim3(B, 4), dim3(256), 0, stream>>>(pred, gt, part, 0);
  k_chamfer<<<dim3(B, 4), dim3(256), 0, stream>>>(gt, pred, part, 32);
  k_loss<<<dim3(1), dim3(1), 0, stream>>>(part, out_loss);
}
